// GraphAutoEncoder_46445776339278
// MI455X (gfx1250) — matrix operand order repaired and checked by reading, not compiled, not run on hardware
//
#include <hip/hip_runtime.h>
#include <hip/hip_bf16.h>
#include <stdint.h>

// ---------------------------------------------------------------------------
// CDNA5 (gfx1250) wave32 WMMA types
// ---------------------------------------------------------------------------
typedef __attribute__((ext_vector_type(16))) __bf16 v16bf;
typedef __attribute__((ext_vector_type(8)))  float  v8f;

__device__ __forceinline__ v8f wmma_bf16(v16bf a, v16bf b, v8f c) {
    // D = A(16x32 bf16) * B(32x16 bf16) + C(16x16 f32)
    return __builtin_amdgcn_wmma_f32_16x16x32_bf16(
        /*neg_a=*/false, a, /*neg_b=*/false, b,
        /*c_mod=*/(short)0, c, /*reuse_a=*/false, /*reuse_b=*/false);
}

// A-fragment: 16x32 bf16 tile, row-major A[M,K] (lda=K).
// Lane L (0..15): row M=L, K = 0..15; lanes 16..31: K = 16..31.
__device__ __forceinline__ v16bf load_a_frag(const __bf16* A, int lda,
                                             int row0, int kb, int lane) {
    int m = row0 + (lane & 15);
    int koff = (lane >= 16) ? 16 : 0;
    const __bf16* p = A + (size_t)m * lda + kb + koff;
    v16bf a;
#pragma unroll
    for (int i = 0; i < 8; ++i)  a[i] = p[i];        // K = koff+0..7
#pragma unroll
    for (int i = 8; i < 16; ++i) a[i] = p[i];        // K = koff+8..15
    return a;
}

// B-fragment from a TRANSPOSED operand Bt[N,K] (so B[k][n] = Bt[n][k]).
// Lane L: col N=L%16; lanes 0..15 hold K=kb..kb+15, lanes 16..31 K=kb+16..kb+31.
// Per-lane loads are contiguous (2 x 16B) -> global_load_b128 pairs.
__device__ __forceinline__ v16bf load_bT_frag(const __bf16* Bt, int ldb,
                                              int kb, int col0, int lane) {
    int n  = col0 + (lane & 15);
    int k0 = kb + ((lane >= 16) ? 16 : 0);
    const __bf16* p = Bt + (size_t)n * ldb + k0;
    v16bf b;
#pragma unroll
    for (int i = 0; i < 16; ++i) b[i] = p[i];
    return b;
}

// C/D 16x16 f32: lane<16 -> (M=v, N=lane); lane>=16 -> (M=v+8, N=lane-16).
__device__ __forceinline__ void store_c(float* C, int ldc, int row0, int col0,
                                        const v8f& c, int lane) {
    int n = col0 + (lane & 15);
    int mbase = row0 + ((lane >= 16) ? 8 : 0);
#pragma unroll
    for (int v = 0; v < 8; ++v)
        C[(size_t)(mbase + v) * ldc + n] = c[v];
}

// ---------------------------------------------------------------------------
// Degree / normalization kernels
// ---------------------------------------------------------------------------
__global__ void deg_init_kernel(float* deg, unsigned* gmax, int n) {
    int i = blockIdx.x * blockDim.x + threadIdx.x;
    if (i < n) deg[i] = 1.0f;            // self-loop
    if (i == 0) *gmax = 0u;
}

__global__ void deg_scatter_kernel(const long long* dst, float* deg, int e) {
    int i = blockIdx.x * blockDim.x + threadIdx.x;
    if (i < e) atomicAdd(&deg[(int)dst[i]], 1.0f);
}

__global__ void deg_finish_kernel(const float* deg, float* dinv, float* invdeg, int n) {
    int i = blockIdx.x * blockDim.x + threadIdx.x;
    if (i < n) {
        float d = deg[i];
        dinv[i]   = rsqrtf(d);
        invdeg[i] = 1.0f / d;
    }
}

// ---------------------------------------------------------------------------
// f32 -> split bf16 (hi + lo) conversion
// ---------------------------------------------------------------------------
__global__ void split_bf16_kernel(const float* X, __bf16* hi, __bf16* lo, int n) {
    int i = blockIdx.x * blockDim.x + threadIdx.x;
    if (i < n) {
        float x = X[i];
        __bf16 h = (__bf16)x;
        hi[i] = h;
        lo[i] = (__bf16)(x - (float)h);
    }
}

// Split + transpose for weight matrices: W[K,N] f32 -> hiT/loT [N,K] bf16.
// Makes all B-fragment loads contiguous (removes strided 16-bit gathers).
__global__ void split_bf16_T_kernel(const float* W, __bf16* hiT, __bf16* loT,
                                    int K, int Nn) {
    int i = blockIdx.x * blockDim.x + threadIdx.x;
    if (i < K * Nn) {
        int k = i / Nn, nn = i - k * Nn;
        float x = W[i];
        __bf16 h = (__bf16)x;
        hiT[(size_t)nn * K + k] = h;
        loT[(size_t)nn * K + k] = (__bf16)(x - (float)h);
    }
}

__global__ void relu_split_kernel(const float* X, __bf16* hi, __bf16* lo, int n) {
    int i = blockIdx.x * blockDim.x + threadIdx.x;
    if (i < n) {
        float x = fmaxf(X[i], 0.0f);
        __bf16 h = (__bf16)x;
        hi[i] = h;
        lo[i] = (__bf16)(x - (float)h);
    }
}

// ---------------------------------------------------------------------------
// Split-bf16 GEMM: C[M,Nn] = (Ah+Al)[M,K] @ (BhT+BlT)^T, B given as [Nn,K].
// (drops lo*lo term; ~f32 accuracy). One wave per 16x16 tile, 4 waves/block.
// ---------------------------------------------------------------------------
__global__ void gemm_bf16x2_kernel(const __bf16* __restrict__ Ah,
                                   const __bf16* __restrict__ Al,
                                   const __bf16* __restrict__ BhT,
                                   const __bf16* __restrict__ BlT,
                                   float* __restrict__ C,
                                   int M, int Nn, int K) {
    int lane = threadIdx.x & 31;
    int w    = threadIdx.x >> 5;
    int tiles_n = Nn >> 4;
    int total   = (M >> 4) * tiles_n;
    int tile = blockIdx.x * 4 + w;          // wave-uniform
    if (tile >= total) return;
    int tm = tile / tiles_n, tn = tile % tiles_n;
    int row0 = tm << 4, col0 = tn << 4;

    v8f acc = {};
    for (int kb = 0; kb < K; kb += 32) {
        v16bf ah = load_a_frag(Ah, K, row0, kb, lane);
        v16bf al = load_a_frag(Al, K, row0, kb, lane);
        v16bf bh = load_bT_frag(BhT, K, kb, col0, lane);
        v16bf bl = load_bT_frag(BlT, K, kb, col0, lane);
        acc = wmma_bf16(ah, bh, acc);
        acc = wmma_bf16(ah, bl, acc);
        acc = wmma_bf16(al, bh, acc);
    }
    store_c(C, Nn, row0, col0, acc, lane);
}

// ---------------------------------------------------------------------------
// GCN aggregation: agg[i] = h[i]/deg[i] + b ; then agg[dst] += h[src]*norm(e)
// ---------------------------------------------------------------------------
__global__ void agg_init_kernel(const float* __restrict__ h,
                                const float* __restrict__ invdeg,
                                const float* __restrict__ bias,
                                float* __restrict__ agg, int n, int d) {
    int i = blockIdx.x * blockDim.x + threadIdx.x;
    int total = n * d;
    if (i < total) {
        int node = i / d, f = i % d;
        agg[i] = h[i] * invdeg[node] + bias[f];
    }
}

__global__ void edge_scatter_kernel(const float* __restrict__ h,
                                    const long long* __restrict__ src,
                                    const long long* __restrict__ dst,
                                    const float* __restrict__ dinv,
                                    float* __restrict__ agg, int e, int d) {
    int t = blockIdx.x * blockDim.x + threadIdx.x;
    int chunks = d >> 2;                       // 4 features per thread
    if (t >= e * chunks) return;
    int ed = t / chunks;
    int f  = (t - ed * chunks) << 2;
    int s  = (int)src[ed];
    int dd = (int)dst[ed];
    float c = dinv[s] * dinv[dd];
    const float4 hv = *(const float4*)(h + (size_t)s * d + f);
    float* o = agg + (size_t)dd * d + f;
    atomicAdd(o + 0, hv.x * c);
    atomicAdd(o + 1, hv.y * c);
    atomicAdd(o + 2, hv.z * c);
    atomicAdd(o + 3, hv.w * c);
}

// ---------------------------------------------------------------------------
// z prep: per-row squared norm + split-bf16 of z
// ---------------------------------------------------------------------------
__global__ void zprep_kernel(const float* __restrict__ Z,
                             __bf16* __restrict__ zh, __bf16* __restrict__ zl,
                             float* __restrict__ sq, int n, int d) {
    int i = blockIdx.x * blockDim.x + threadIdx.x;
    if (i >= n) return;
    const float* row = Z + (size_t)i * d;
    float s = 0.0f;
#pragma unroll 8
    for (int f = 0; f < d; ++f) {
        float x = row[f];
        s += x * x;
        __bf16 h = (__bf16)x;
        zh[(size_t)i * d + f] = h;
        zl[(size_t)i * d + f] = (__bf16)(x - (float)h);
    }
    sq[i] = s;
}

// ---------------------------------------------------------------------------
// Decode: dist[i,j] = sqrt(max(sq_i + sq_j - 2*z_i.z_j, 0)); track global max.
// z @ z^T via split-bf16 WMMA. Each wave computes a 16x32 strip (two 16x16
// tiles sharing A-fragments) so row stores cover 128B across the wave.
// K = 64 = 2 k-steps x 3 wmma x 2 tiles.
// ---------------------------------------------------------------------------
__global__ void decode_kernel(const __bf16* __restrict__ Zh,
                              const __bf16* __restrict__ Zl,
                              const float* __restrict__ sq,
                              float* __restrict__ out,
                              unsigned* __restrict__ gmax,
                              int n, int d, int tiles, int pairs) {
    __shared__ float smax[4];
    int lane = threadIdx.x & 31;
    int w    = threadIdx.x >> 5;
    int tile = blockIdx.x * 4 + w;          // wave-uniform
    int total = tiles * pairs;
    float lmax = 0.0f;

    if (tile < total) {
        int tm = tile / pairs, tp = tile % pairs;
        int row0 = tm << 4, col0 = tp << 5;
        bool has2 = (col0 + 16) < n;         // wave-uniform tail guard

        v8f acc0 = {};
        v8f acc1 = {};
        for (int kb = 0; kb < d; kb += 32) {
            v16bf ah = load_a_frag(Zh, d, row0, kb, lane);
            v16bf al = load_a_frag(Zl, d, row0, kb, lane);
            v16bf bh0 = load_bT_frag(Zh, d, kb, col0, lane);
            v16bf bl0 = load_bT_frag(Zl, d, kb, col0, lane);
            acc0 = wmma_bf16(ah, bh0, acc0);
            acc0 = wmma_bf16(ah, bl0, acc0);
            acc0 = wmma_bf16(al, bh0, acc0);
            if (has2) {
                v16bf bh1 = load_bT_frag(Zh, d, kb, col0 + 16, lane);
                v16bf bl1 = load_bT_frag(Zl, d, kb, col0 + 16, lane);
                acc1 = wmma_bf16(ah, bh1, acc1);
                acc1 = wmma_bf16(ah, bl1, acc1);
                acc1 = wmma_bf16(al, bh1, acc1);
            }
        }

        int nn0   = col0 + (lane & 15);
        int mbase = row0 + ((lane >= 16) ? 8 : 0);
        float sqn0 = sq[nn0];
        float sqn1 = has2 ? sq[nn0 + 16] : 0.0f;
#pragma unroll
        for (int v = 0; v < 8; ++v) {
            int mm = mbase + v;
            float sqm = sq[mm];
            float* rowp = out + (size_t)mm * n;

            float d2 = fmaxf(sqm + sqn0 - 2.0f * acc0[v], 0.0f);
            float dist = __builtin_sqrtf(d2);
            rowp[nn0] = dist;
            lmax = fmaxf(lmax, dist);

            if (has2) {
                float e2 = fmaxf(sqm + sqn1 - 2.0f * acc1[v], 0.0f);
                float dist1 = __builtin_sqrtf(e2);
                rowp[nn0 + 16] = dist1;
                lmax = fmaxf(lmax, dist1);
            }
        }
    }

    // wave32 reduction then block reduction; single atomicMax per block.
#pragma unroll
    for (int off = 16; off > 0; off >>= 1)
        lmax = fmaxf(lmax, __shfl_xor(lmax, off, 32));
    if (lane == 0) smax[w] = lmax;
    __syncthreads();
    if (threadIdx.x == 0) {
        float m = fmaxf(fmaxf(smax[0], smax[1]), fmaxf(smax[2], smax[3]));
        atomicMax(gmax, __float_as_uint(m));   // valid: dist >= 0
    }
}

// ---------------------------------------------------------------------------
// Rescale: out = 1 - dist / max  (vectorized float4, grid-stride)
// ---------------------------------------------------------------------------
__global__ void rescale_kernel(float* __restrict__ out,
                               const unsigned* __restrict__ gmax,
                               size_t total4) {
    float m = __uint_as_float(*gmax);
    if (m == 0.0f) m = 1.0f;
    float inv = 1.0f / m;
    float4* o4 = (float4*)out;
    size_t stride = (size_t)gridDim.x * blockDim.x;
    for (size_t i = (size_t)blockIdx.x * blockDim.x + threadIdx.x;
         i < total4; i += stride) {
        float4 v = o4[i];
        v.x = 1.0f - v.x * inv;
        v.y = 1.0f - v.y * inv;
        v.z = 1.0f - v.z * inv;
        v.w = 1.0f - v.w * inv;
        o4[i] = v;
    }
}

// ---------------------------------------------------------------------------
// Host launch
// ---------------------------------------------------------------------------
static inline char* bump(char*& p, size_t bytes) {
    char* r = p;
    p += (bytes + 255) & ~(size_t)255;
    return r;
}
static inline int cdiv(int a, int b) { return (a + b - 1) / b; }

extern "C" void kernel_launch(void* const* d_in, const int* in_sizes, int n_in,
                              void* d_out, int out_size, void* d_ws, size_t ws_size,
                              hipStream_t stream) {
    const int DIN = 128, DH = 256, DOUT = 64;

    const float*     x   = (const float*)d_in[0];
    const long long* ei  = (const long long*)d_in[1];
    const float*     W1  = (const float*)d_in[2];
    const float*     b1  = (const float*)d_in[3];
    const float*     W2  = (const float*)d_in[4];
    const float*     b2  = (const float*)d_in[5];
    float* out = (float*)d_out;

    const int N = in_sizes[0] / DIN;
    const int E = in_sizes[1] / 2;
    const long long* srcIdx = ei;
    const long long* dstIdx = ei + E;

    // workspace layout
    char* p = (char*)d_ws;
    float*    deg    = (float*)   bump(p, (size_t)N * 4);
    float*    dinv   = (float*)   bump(p, (size_t)N * 4);
    float*    invdeg = (float*)   bump(p, (size_t)N * 4);
    unsigned* gmax   = (unsigned*)bump(p, 256);
    __bf16* x_hi   = (__bf16*)bump(p, (size_t)N * DIN * 2);
    __bf16* x_lo   = (__bf16*)bump(p, (size_t)N * DIN * 2);
    __bf16* w1t_hi = (__bf16*)bump(p, (size_t)DIN * DH * 2);   // [DH, DIN]
    __bf16* w1t_lo = (__bf16*)bump(p, (size_t)DIN * DH * 2);
    __bf16* w2t_hi = (__bf16*)bump(p, (size_t)DH * DOUT * 2);  // [DOUT, DH]
    __bf16* w2t_lo = (__bf16*)bump(p, (size_t)DH * DOUT * 2);
    float*  hx1    = (float*) bump(p, (size_t)N * DH * 4);
    float*  agg1   = (float*) bump(p, (size_t)N * DH * 4);
    __bf16* h1_hi  = (__bf16*)bump(p, (size_t)N * DH * 2);
    __bf16* h1_lo  = (__bf16*)bump(p, (size_t)N * DH * 2);
    float*  hx2    = (float*) bump(p, (size_t)N * DOUT * 4);
    float*  z      = (float*) bump(p, (size_t)N * DOUT * 4);
    __bf16* z_hi   = (__bf16*)bump(p, (size_t)N * DOUT * 2);
    __bf16* z_lo   = (__bf16*)bump(p, (size_t)N * DOUT * 2);
    float*  sq     = (float*) bump(p, (size_t)N * 4);

    const int T = 256;

    // 1. degrees / normalization
    deg_init_kernel   <<<cdiv(N, T), T, 0, stream>>>(deg, gmax, N);
    deg_scatter_kernel<<<cdiv(E, T), T, 0, stream>>>(dstIdx, deg, E);
    deg_finish_kernel <<<cdiv(N, T), T, 0, stream>>>(deg, dinv, invdeg, N);

    // 2. split inputs to bf16 hi/lo; weights split + transposed
    split_bf16_kernel  <<<cdiv(N * DIN, T),   T, 0, stream>>>(x,  x_hi,  x_lo,  N * DIN);
    split_bf16_T_kernel<<<cdiv(DIN * DH, T),  T, 0, stream>>>(W1, w1t_hi, w1t_lo, DIN, DH);
    split_bf16_T_kernel<<<cdiv(DH * DOUT, T), T, 0, stream>>>(W2, w2t_hi, w2t_lo, DH, DOUT);

    // 3. layer 1: hx1 = x @ W1 (WMMA), agg, relu
    {
        int tiles = (N >> 4) * (DH >> 4);
        gemm_bf16x2_kernel<<<cdiv(tiles, 4), 128, 0, stream>>>(
            x_hi, x_lo, w1t_hi, w1t_lo, hx1, N, DH, DIN);
    }
    agg_init_kernel<<<cdiv(N * DH, T), T, 0, stream>>>(hx1, invdeg, b1, agg1, N, DH);
    edge_scatter_kernel<<<cdiv(E * (DH >> 2), T), T, 0, stream>>>(
        hx1, srcIdx, dstIdx, dinv, agg1, E, DH);
    relu_split_kernel<<<cdiv(N * DH, T), T, 0, stream>>>(agg1, h1_hi, h1_lo, N * DH);

    // 4. layer 2: hx2 = h1 @ W2 (WMMA), agg -> z
    {
        int tiles = (N >> 4) * (DOUT >> 4);
        gemm_bf16x2_kernel<<<cdiv(tiles, 4), 128, 0, stream>>>(
            h1_hi, h1_lo, w2t_hi, w2t_lo, hx2, N, DOUT, DH);
    }
    agg_init_kernel<<<cdiv(N * DOUT, T), T, 0, stream>>>(hx2, invdeg, b2, z, N, DOUT);
    edge_scatter_kernel<<<cdiv(E * (DOUT >> 2), T), T, 0, stream>>>(
        hx2, srcIdx, dstIdx, dinv, z, E, DOUT);

    // 5. decode prep + z @ z^T distance matrix (WMMA) + max
    zprep_kernel<<<cdiv(N, T), T, 0, stream>>>(z, z_hi, z_lo, sq, N, DOUT);
    {
        int tiles = N >> 4;                 // 625 row tiles
        int pairs = cdiv(tiles, 2);         // 313 column strips of 32
        int total = tiles * pairs;          // wave-strips
        decode_kernel<<<cdiv(total, 4), 128, 0, stream>>>(
            z_hi, z_lo, sq, out, gmax, N, DOUT, tiles, pairs);
    }

    // 6. normalize: out = 1 - dist / max
    {
        size_t total4 = ((size_t)N * N) / 4;
        rescale_kernel<<<4096, 256, 0, stream>>>(out, gmax, total4);
    }
}
